// ISAPVLADV2_7121055776894
// MI455X (gfx1250) — matrix operand order repaired and checked by reading, not compiled, not run on hardware
//
#include <hip/hip_runtime.h>
#include <math.h>
#include <stdint.h>

typedef unsigned short u16;
typedef __attribute__((ext_vector_type(16))) __bf16 v16bf;
typedef __attribute__((ext_vector_type(8)))  float  v8f;
typedef int v4i __attribute__((vector_size(16)));   // type expected by async-LDS builtin

union FragAB {
  v16bf v;
  u16   u[16];
  uint4 q[2];
};

__device__ __forceinline__ u16 f2bf(float f) {
  unsigned int x = __float_as_uint(f);
  unsigned int r = x + 0x7FFFu + ((x >> 16) & 1u);
  return (u16)(r >> 16);
}

static constexpr int N_ = 32, C_ = 128, H_ = 48, W_ = 48, K_ = 64, S_ = 4, L_ = 3;
static constexpr int P_ = H_ * W_;   // 2304
static constexpr int ROWS = 512;     // 64 score + 256 shadow + 192 app
static constexpr int R_ = 21;        // 1 + 4 + 16 regions

// ---------------------------------------------------------------------------
// Kernel 0: cast fused weight matrix [512,128] to bf16
// rows 0..63 = conv_w, 64..319 = conv_shadow_w (k*S+s), 320..511 = app (l*64+k)
// ---------------------------------------------------------------------------
__global__ __launch_bounds__(256) void prep_kernel(
    const float* __restrict__ cw, const float* __restrict__ sw,
    const float* __restrict__ apw, u16* __restrict__ Wb) {
  int idx = blockIdx.x * 256 + threadIdx.x;   // 0..65535
  float v;
  if (idx < 64 * 128)            v = cw[idx];
  else if (idx < 320 * 128)      v = sw[idx - 64 * 128];
  else                           v = apw[idx - 320 * 128];
  Wb[idx] = f2bf(v);
}

// ---------------------------------------------------------------------------
// Kernel 1: per 16-pixel tile: L2-normalize x, cast to bf16 (store [p][c]),
// fused 512x16 WMMA GEMM against Wb, then per-pixel softmax + shadow gating
// + pyramid ReLU region partial sums (atomics into LDS, then global).
// ---------------------------------------------------------------------------
__global__ __launch_bounds__(256) void assign_kernel(
    const float* __restrict__ x, const u16* __restrict__ Wb,
    u16* __restrict__ xbg, float* __restrict__ a_glob,
    float* __restrict__ sal_glob)
{
  __shared__ float xtile[16 * 128];     // f32, [pixel][c]
  __shared__ u16   xbl[16 * 128];       // bf16 normalized, [pixel][c]
  __shared__ float scoreL[ROWS * 16];   // [row][pixel]
  __shared__ float salL[K_ * R_];
  __shared__ float rn[16];

  const int t  = threadIdx.x;
  const int pt = blockIdx.x;   // 0..143 pixel tile
  const int n  = blockIdx.y;   // 0..31 batch
  const int p0 = pt * 16;

  for (int i = t; i < K_ * R_; i += 256) salL[i] = 0.0f;

  // load x tile (N,C,P layout), transpose to [pi][c]
  {
    const float* xb = x + (size_t)n * C_ * P_ + p0;
    #pragma unroll
    for (int pass = 0; pass < 8; ++pass) {
      int idx = pass * 256 + t;          // 0..2047
      int c = idx >> 4, pi = idx & 15;
      xtile[pi * 128 + c] = xb[(size_t)c * P_ + pi];
    }
  }
  __syncthreads();
  if (t < 16) {
    float s = 0.0f;
    #pragma unroll 4
    for (int c = 0; c < 128; ++c) { float v = xtile[t * 128 + c]; s += v * v; }
    rn[t] = 1.0f / fmaxf(sqrtf(s), 1e-12f);
  }
  __syncthreads();
  {
    u16* xg = xbg + (size_t)n * P_ * C_ + (size_t)p0 * C_;
    #pragma unroll
    for (int pass = 0; pass < 8; ++pass) {
      int idx = pass * 256 + t;
      u16 b = f2bf(xtile[idx] * rn[idx >> 7]);
      xbl[idx] = b;
      xg[idx]  = b;                      // contiguous -> coalesced
    }
  }
  __syncthreads();

  // ---- WMMA GEMM: scoreL[512][16] = Wb[512][128] * xbl^T[128][16] ----
  {
    const int wv = t >> 5, lane = t & 31, lr = lane & 15, hi = lane >> 4;
    v8f acc[4];
    const v8f vz = {0.f,0.f,0.f,0.f,0.f,0.f,0.f,0.f};
    acc[0] = vz; acc[1] = vz; acc[2] = vz; acc[3] = vz;
    #pragma unroll
    for (int kk = 0; kk < 4; ++kk) {            // K = 128 in chunks of 32
      FragAB bf;
      const uint4* bq = (const uint4*)&xbl[lr * 128 + kk * 32];
      bf.q[0] = bq[hi]; bf.q[1] = bq[2 + hi];
      #pragma unroll
      for (int t4 = 0; t4 < 4; ++t4) {          // 4 row tiles per wave
        const u16* wr = Wb + (size_t)(wv * 64 + t4 * 16 + lr) * 128 + kk * 32;
        FragAB af;
        af.q[0] = *(const uint4*)(wr + hi * 8);
        af.q[1] = *(const uint4*)(wr + 16 + hi * 8);
        acc[t4] = __builtin_amdgcn_wmma_f32_16x16x32_bf16(
            false, af.v, false, bf.v, (short)0, acc[t4], false, false);
      }
    }
    #pragma unroll
    for (int t4 = 0; t4 < 4; ++t4)
      #pragma unroll
      for (int v = 0; v < 8; ++v)
        scoreL[(wv * 64 + t4 * 16 + v + hi * 8) * 16 + lr] = acc[t4][v];
  }
  __syncthreads();

  // ---- per-pixel: softmax over K, shadow softmax, region partials ----
  {
    const int pi = t >> 4;           // pixel within tile
    const int kb = (t & 15) * 4;     // 4 clusters per thread
    const int pg = p0 + pi;
    const int h = pg / W_, w = pg % W_;

    float mx = -3.4e38f;
    for (int k = 0; k < K_; ++k) mx = fmaxf(mx, scoreL[k * 16 + pi]);
    float den = 0.0f;
    for (int k = 0; k < K_; ++k) den += __expf(scoreL[k * 16 + pi] - mx);

    const int NS[3] = {1,2,4}, WID[3] = {48,32,20}, STR[3] = {24,16,10},
              PAD[3] = {0,0,1}, RB[3] = {0,1,5};

    for (int kk = 0; kk < 4; ++kk) {
      int k = kb + kk;
      float s = scoreL[k * 16 + pi];
      float ak = __expf(s - mx) / den;
      float m2 = s;
      #pragma unroll
      for (int si = 0; si < S_; ++si)
        m2 = fmaxf(m2, scoreL[(64 + k * S_ + si) * 16 + pi]);
      float d2 = __expf(s - m2);
      #pragma unroll
      for (int si = 0; si < S_; ++si)
        d2 += __expf(scoreL[(64 + k * S_ + si) * 16 + pi] - m2);
      ak *= __expf(s - m2) / d2;
      a_glob[((size_t)n * K_ + k) * P_ + pg] = ak;

      #pragma unroll
      for (int l = 0; l < L_; ++l) {
        float ap = scoreL[(320 + l * 64 + k) * 16 + pi];
        if (ap > 0.0f) {
          for (int i = 0; i < NS[l]; ++i) {
            int hs = i * STR[l] - PAD[l];
            if (h < hs || h >= hs + WID[l]) continue;
            for (int j = 0; j < NS[l]; ++j) {
              int ws = j * STR[l] - PAD[l];
              if (w < ws || w >= ws + WID[l]) continue;
              atomicAdd(&salL[k * R_ + RB[l] + i * NS[l] + j], ap);
            }
          }
        }
      }
    }
  }
  __syncthreads();
  for (int i = t; i < K_ * R_; i += 256)
    atomicAdd(&sal_glob[(size_t)n * K_ * R_ + i], salL[i]);
}

// ---------------------------------------------------------------------------
// Kernel 2: per (n,k): w[p] = sum_r M[r,p]*sal[r];  aw = a*w (bf16); awsum
// ---------------------------------------------------------------------------
__global__ __launch_bounds__(256) void aw_kernel(
    const float* __restrict__ a_glob, const float* __restrict__ sal_glob,
    u16* __restrict__ awb, float* __restrict__ awsum)
{
  const int k = blockIdx.x, n = blockIdx.y, t = threadIdx.x;
  __shared__ float ssal[R_];
  __shared__ float red[256];
  if (t < R_) ssal[t] = sal_glob[((size_t)n * K_ + k) * R_ + t];
  __syncthreads();
  const int NS[3]={1,2,4}, WID[3]={48,32,20}, STR[3]={24,16,10},
            PAD[3]={0,0,1}, RB[3]={0,1,5};
  float acc = 0.0f;
  for (int p = t; p < P_; p += 256) {
    int h = p / W_, w = p % W_;
    float wg = 0.0f;
    #pragma unroll
    for (int l = 0; l < L_; ++l)
      for (int i = 0; i < NS[l]; ++i) {
        int hs = i * STR[l] - PAD[l];
        if (h < hs || h >= hs + WID[l]) continue;
        for (int j = 0; j < NS[l]; ++j) {
          int ws = j * STR[l] - PAD[l];
          if (w < ws || w >= ws + WID[l]) continue;
          wg += ssal[RB[l] + i * NS[l] + j];
        }
      }
    float av = a_glob[((size_t)n * K_ + k) * P_ + p] * wg;
    awb[((size_t)n * K_ + k) * P_ + p] = f2bf(av);
    acc += av;
  }
  red[t] = acc; __syncthreads();
  for (int off = 128; off > 0; off >>= 1) {
    if (t < off) red[t] += red[t + off];
    __syncthreads();
  }
  if (t == 0) awsum[n * K_ + k] = red[0];
}

// ---------------------------------------------------------------------------
// Kernel 3: vlad[n][64][128] = aw[64][P] @ xnorm^T[P][128] - centroid*awsum
// WMMA over P in chunks of 32; x chunk DMA'd into LDS via async-to-LDS ops
// (ASYNCcnt path) with a prefetch of the next chunk.
// ---------------------------------------------------------------------------
__global__ __launch_bounds__(256) void vlad_gemm_kernel(
    const u16* __restrict__ awb, const u16* __restrict__ xbg,
    const float* __restrict__ centroids, const float* __restrict__ awsum,
    float* __restrict__ vlad)
{
  const int n = blockIdx.x, t = threadIdx.x;
  const int wv = t >> 5, lane = t & 31, lr = lane & 15, hi = lane >> 4;
  const int mt = wv >> 1, ntb = (wv & 1) * 4;   // wave -> 1 M-tile x 4 N-tiles
  __shared__ u16 xch[32 * 128];                 // [p][c] chunk (8 KB)
  v8f acc[4];
  const v8f vz = {0.f,0.f,0.f,0.f,0.f,0.f,0.f,0.f};
  acc[0] = vz; acc[1] = vz; acc[2] = vz; acc[3] = vz;

  const char* xbase = (const char*)(xbg + (size_t)n * P_ * C_);

  for (int pc = 0; pc < P_ / 32; ++pc) {
    __syncthreads();
    {
      const char* src = xbase + (size_t)pc * 32 * C_ * 2 + t * 32;  // 32 B/thread
      char*       dst = (char*)xch + t * 32;
#if __has_builtin(__builtin_amdgcn_global_load_async_to_lds_b128)
      __builtin_amdgcn_global_load_async_to_lds_b128((v4i*)src, (v4i*)dst, 0, 0);
      __builtin_amdgcn_global_load_async_to_lds_b128((v4i*)src, (v4i*)dst, 16, 0);
      if (pc + 1 < P_ / 32) __builtin_prefetch(src + 32 * C_ * 2, 0, 0);
      asm volatile("s_wait_asynccnt 0" ::: "memory");
#else
      ((uint4*)dst)[0] = ((const uint4*)src)[0];
      ((uint4*)dst)[1] = ((const uint4*)src)[1];
#endif
    }
    __syncthreads();
    const u16* ar = awb + ((size_t)n * K_ + mt * 16 + lr) * P_ + pc * 32;
    FragAB af;
    af.q[0] = *(const uint4*)(ar + hi * 8);
    af.q[1] = *(const uint4*)(ar + 16 + hi * 8);
    #pragma unroll
    for (int nt = 0; nt < 4; ++nt) {
      const int c0 = (ntb + nt) * 16;
      FragAB bf;
      #pragma unroll
      for (int i = 0; i < 16; ++i)
        bf.u[i] = xch[(hi * 8 + i + (i & 8)) * 128 + c0 + lr];
      acc[nt] = __builtin_amdgcn_wmma_f32_16x16x32_bf16(
          false, af.v, false, bf.v, (short)0, acc[nt], false, false);
    }
  }
  #pragma unroll
  for (int nt = 0; nt < 4; ++nt) {
    int c = (ntb + nt) * 16 + lr;
    #pragma unroll
    for (int v = 0; v < 8; ++v) {
      int k = mt * 16 + v + hi * 8;
      float val = acc[nt][v] - centroids[k * 128 + c] * awsum[n * K_ + k];
      vlad[((size_t)n * K_ + k) * C_ + c] = val;
    }
  }
}

// ---------------------------------------------------------------------------
// Kernel 4: intra-norm per (k), * cluster_weights, global L2 over 8192, store
// ---------------------------------------------------------------------------
__global__ __launch_bounds__(256) void finalize_kernel(
    const float* __restrict__ vlad, const float* __restrict__ cwts,
    float* __restrict__ out)
{
  const int n = blockIdx.x, t = threadIdx.x;
  const int k = t >> 2, q = t & 3;
  __shared__ float pk[256];
  __shared__ float scale[K_];
  __shared__ float tot[256];
  const float* row = vlad + ((size_t)n * K_ + k) * C_;
  float s = 0.0f;
  for (int c = q * 32; c < q * 32 + 32; ++c) { float v = row[c]; s += v * v; }
  pk[t] = s; __syncthreads();
  if (q == 0) {
    float nr = sqrtf(pk[t] + pk[t + 1] + pk[t + 2] + pk[t + 3]);
    scale[k] = cwts[k] / fmaxf(nr, 1e-12f);
  }
  __syncthreads();
  float ts = 0.0f;
  for (int c = q * 32; c < q * 32 + 32; ++c) { float v = row[c] * scale[k]; ts += v * v; }
  tot[t] = ts; __syncthreads();
  for (int off = 128; off > 0; off >>= 1) {
    if (t < off) tot[t] += tot[t + off];
    __syncthreads();
  }
  float inv = 1.0f / fmaxf(sqrtf(tot[0]), 1e-12f);
  for (int c = q * 32; c < q * 32 + 32; ++c)
    out[(size_t)n * (K_ * C_) + k * C_ + c] = row[c] * scale[k] * inv;
}

// ---------------------------------------------------------------------------
extern "C" void kernel_launch(void* const* d_in, const int* in_sizes, int n_in,
                              void* d_out, int out_size, void* d_ws, size_t ws_size,
                              hipStream_t stream) {
  (void)in_sizes; (void)n_in; (void)out_size; (void)ws_size;
  const float* x     = (const float*)d_in[0];
  const float* cent  = (const float*)d_in[1];
  const float* convw = (const float*)d_in[2];
  const float* shadw = (const float*)d_in[3];
  const float* appw  = (const float*)d_in[4];
  const float* cwts  = (const float*)d_in[5];

  char* w = (char*)d_ws;
  u16*   Wb     = (u16*)w;   w += (size_t)ROWS * C_ * 2;            // 128 KB
  u16*   xbg    = (u16*)w;   w += (size_t)N_ * P_ * C_ * 2;         // 18.9 MB
  float* a_glob = (float*)w; w += (size_t)N_ * K_ * P_ * 4;         // 18.9 MB
  float* sal    = (float*)w; w += (size_t)N_ * K_ * R_ * 4;         // 172 KB
  u16*   awb    = (u16*)w;   w += (size_t)N_ * K_ * P_ * 2;         // 9.4 MB
  float* awsum  = (float*)w; w += (size_t)N_ * K_ * 4;              // 8 KB
  float* vlad   = (float*)w; w += (size_t)N_ * K_ * C_ * 4;         // 1.0 MB

  (void)hipMemsetAsync(sal, 0, (size_t)N_ * K_ * R_ * 4, stream);

  prep_kernel<<<256, 256, 0, stream>>>(convw, shadw, appw, Wb);
  assign_kernel<<<dim3(P_ / 16, N_), 256, 0, stream>>>(x, Wb, xbg, a_glob, sal);
  aw_kernel<<<dim3(K_, N_), 256, 0, stream>>>(a_glob, sal, awb, awsum);
  vlad_gemm_kernel<<<N_, 256, 0, stream>>>(awb, xbg, cent, awsum, vlad);
  finalize_kernel<<<N_, 256, 0, stream>>>(vlad, cwts, (float*)d_out);
}
